// RGCNAttnConv_32573031973189
// MI455X (gfx1250) — matrix operand order repaired and checked by reading, not compiled, not run on hardware
//
#include <hip/hip_runtime.h>

// ---------------------------------------------------------------------------
// RGCN + attention conv for MI455X (gfx1250, wave32, WMMA).
// Heavy GEMMs (per-edge message x[src]@W[r], proj, root) run on
// v_wmma_f32_16x16x32_f16. Edges are counting-sorted by relation into
// 16-edge tiles so each wave does a dense WMMA tile with one W[r].
// Softmax uses order-preserving-uint atomicMax + fp32 atomicAdd segments.
// ---------------------------------------------------------------------------

typedef __attribute__((ext_vector_type(16))) _Float16 v16h;
typedef __attribute__((ext_vector_type(8)))  float    v8f;

#define N_NODES   100000
#define N_EDGES   1000000
#define FIN       64
#define FOUT      32
#define NH        4
#define NR        8
#define HF        128          /* NH*FOUT */
#define NEG_SLOPE 0.2f
#define EPAD      (N_EDGES + NR * 16)
#define NTILES    (EPAD / 16)

// ---- helpers ---------------------------------------------------------------

__device__ __forceinline__ unsigned fkey(float f) {
  unsigned u = __float_as_uint(f);
  return (u & 0x80000000u) ? ~u : (u | 0x80000000u);   // order-preserving
}
__device__ __forceinline__ float funkey(unsigned k) {
  return __uint_as_float((k & 0x80000000u) ? (k & 0x7FFFFFFFu) : ~k);
}
__device__ __forceinline__ float leaky(float v) {
  return v > 0.f ? v : NEG_SLOPE * v;
}

// A fragment: 16x32 f16 tile, row-major source row of length 64 (2 k-steps).
// lane row = lane%16; half-group lh = lane/16:
//   halves 0..7  -> K = ks*32 + lh*16 + j
//   halves 8..15 -> K = ks*32 + lh*16 + 8 + j
__device__ __forceinline__ v16h load_a16(const _Float16* __restrict__ row,
                                         int ks, int lh) {
  v16h a;
  const int b = ks * 32 + lh * 16;
#pragma unroll
  for (int j = 0; j < 8; ++j) {
    a[j]     = row[b + j];
    a[j + 8] = row[b + 8 + j];
  }
  return a;
}

// B fragment: 32x16 f16 tile from a K-major (transposed) column of length 64.
// lane col = lane%16; halves 0..15 -> K = ks*32 + lh*16 + j
__device__ __forceinline__ v16h load_b16(const _Float16* __restrict__ col,
                                         int ks, int lh) {
  v16h b;
  const int base = ks * 32 + lh * 16;
#pragma unroll
  for (int j = 0; j < 16; ++j) b[j] = col[base + j];
  return b;
}

// ---- init / prep -----------------------------------------------------------

__global__ void k_init(unsigned* __restrict__ segmax, float* __restrict__ segsum,
                       int* __restrict__ sorted, int* __restrict__ counts,
                       int* __restrict__ cursors) {
  long long i = (long long)blockIdx.x * blockDim.x + threadIdx.x;
  const long long tot = (long long)NR * N_NODES * NH;
  if (i < tot) { segmax[i] = 0u; segsum[i] = 0.f; }   // key 0 < key(-inf)
  if (i < EPAD) sorted[i] = -1;
  if (i < NR) { counts[i] = 0; cursors[i] = 0; }
}

__global__ void k_xh(const float* __restrict__ x, _Float16* __restrict__ xh) {
  long long i = (long long)blockIdx.x * blockDim.x + threadIdx.x;
  if (i < (long long)N_NODES * FIN) xh[i] = (_Float16)x[i];
}

// W[r][h][i][o] -> Wht[r][h*32+o][i]  (K-major B for message GEMM)
__global__ void k_wt(const float* __restrict__ W, _Float16* __restrict__ wht) {
  int t = blockIdx.x * blockDim.x + threadIdx.x;
  if (t >= NR * NH * FIN * FOUT) return;
  int r = t >> 13, rem = t & 8191;
  int h = rem >> 11, rem2 = rem & 2047;
  int i = rem2 >> 5, o = rem2 & 31;
  wht[((r * HF) + h * FOUT + o) * FIN + i] = (_Float16)W[t];
}

// uht[c][k]: c<32 -> u_s[rh=c][k], c>=32 -> u_d[rh=c-32][k]; K-major.
__global__ void k_u(const float* __restrict__ W, const float* __restrict__ as,
                    const float* __restrict__ ad, _Float16* __restrict__ uht) {
  int t = blockIdx.x * blockDim.x + threadIdx.x;
  if (t >= 64 * 64) return;
  int c = t >> 6, k = t & 63;
  int rh = c & 31;
  const float* att = (c < 32) ? as : ad;
  const float* wrow = W + ((long long)rh * FIN + k) * FOUT; // W[r][h][k][:]
  const float* arow = att + rh * FOUT;
  float s = 0.f;
#pragma unroll
  for (int o = 0; o < FOUT; ++o) s += wrow[o] * arow[o];
  uht[c * 64 + k] = (_Float16)s;
}

// W_root[k][c] -> wrt[c][k]
__global__ void k_wrt(const float* __restrict__ wroot, _Float16* __restrict__ wrt) {
  int t = blockIdx.x * blockDim.x + threadIdx.x;
  if (t >= HF * FIN) return;
  int c = t >> 6, k = t & 63;
  wrt[t] = (_Float16)wroot[k * HF + c];
}

// ---- proj GEMM: proj[n][0..31]=x@u_s, [32..63]=x@u_d -----------------------

__global__ void k_proj(const _Float16* __restrict__ xh,
                       const _Float16* __restrict__ uht,
                       float* __restrict__ proj) {
  const int wave = threadIdx.x >> 5, lane = threadIdx.x & 31;
  const int lh = lane >> 4, lr = lane & 15;
  const int tile = blockIdx.x * (blockDim.x >> 5) + wave;
  if (tile * 16 >= N_NODES) return;
  const _Float16* row = xh + (long long)(tile * 16 + lr) * FIN;
  v16h a0 = load_a16(row, 0, lh);
  v16h a1 = load_a16(row, 1, lh);
#pragma unroll
  for (int c0 = 0; c0 < 64; c0 += 16) {
    v8f acc = {};
    const _Float16* col = uht + (c0 + lr) * 64;
    acc = __builtin_amdgcn_wmma_f32_16x16x32_f16(false, a0, false,
            load_b16(col, 0, lh), (short)0, acc, false, false);
    acc = __builtin_amdgcn_wmma_f32_16x16x32_f16(false, a1, false,
            load_b16(col, 1, lh), (short)0, acc, false, false);
#pragma unroll
    for (int v = 0; v < 8; ++v) {
      int m = v + 8 * lh;
      proj[(long long)(tile * 16 + m) * 64 + c0 + lr] = acc[v];
    }
  }
}

// ---- segment softmax pieces ------------------------------------------------

__global__ void k_segmax(const int* __restrict__ esrc, const int* __restrict__ edst,
                         const int* __restrict__ etyp, const float* __restrict__ proj,
                         unsigned* __restrict__ segmax) {
  long long t = (long long)blockIdx.x * blockDim.x + threadIdx.x;
  if (t >= (long long)N_EDGES * NH) return;
  int e = (int)(t >> 2), h = (int)(t & 3);
  int s = esrc[e], d = edst[e], r = etyp[e];
  float lg = leaky(proj[(long long)s * 64 + r * 4 + h] +
                   proj[(long long)d * 64 + 32 + r * 4 + h]);
  atomicMax(&segmax[((long long)r * N_NODES + d) * 4 + h], fkey(lg));
}

__global__ void k_segsum(const int* __restrict__ esrc, const int* __restrict__ edst,
                         const int* __restrict__ etyp, const float* __restrict__ proj,
                         const unsigned* __restrict__ segmax,
                         float* __restrict__ segsum) {
  long long t = (long long)blockIdx.x * blockDim.x + threadIdx.x;
  if (t >= (long long)N_EDGES * NH) return;
  int e = (int)(t >> 2), h = (int)(t & 3);
  int s = esrc[e], d = edst[e], r = etyp[e];
  long long seg = ((long long)r * N_NODES + d) * 4 + h;
  float lg = leaky(proj[(long long)s * 64 + r * 4 + h] +
                   proj[(long long)d * 64 + 32 + r * 4 + h]);
  atomicAdd(&segsum[seg], __expf(lg - funkey(segmax[seg])));
}

// ---- counting sort by relation --------------------------------------------

__global__ void k_hist(const int* __restrict__ etyp, int* __restrict__ counts) {
  int i = blockIdx.x * blockDim.x + threadIdx.x;
  if (i < N_EDGES) atomicAdd(&counts[etyp[i]], 1);
}

__global__ void k_scan(const int* __restrict__ counts, int* __restrict__ offs) {
  if (blockIdx.x == 0 && threadIdx.x == 0) {
    int o = 0;
    for (int r = 0; r < NR; ++r) { offs[r] = o; o = (o + counts[r] + 15) & ~15; }
    offs[NR] = o;
  }
}

__global__ void k_scatter(const int* __restrict__ etyp, const int* __restrict__ offs,
                          int* __restrict__ cursors, int* __restrict__ sorted) {
  int i = blockIdx.x * blockDim.x + threadIdx.x;
  if (i >= N_EDGES) return;
  int r = etyp[i];
  int p = offs[r] + atomicAdd(&cursors[r], 1);
  sorted[p] = i;
}

// ---- root GEMM: out = x @ W_root + bias -----------------------------------

__global__ void k_root(const _Float16* __restrict__ xh,
                       const _Float16* __restrict__ wrt,
                       const float* __restrict__ bias, float* __restrict__ out) {
  const int wave = threadIdx.x >> 5, lane = threadIdx.x & 31;
  const int lh = lane >> 4, lr = lane & 15;
  const int tile = blockIdx.x * (blockDim.x >> 5) + wave;
  if (tile * 16 >= N_NODES) return;
  const _Float16* row = xh + (long long)(tile * 16 + lr) * FIN;
  v16h a0 = load_a16(row, 0, lh);
  v16h a1 = load_a16(row, 1, lh);
#pragma unroll
  for (int c0 = 0; c0 < HF; c0 += 16) {
    v8f acc = {};
    const _Float16* col = wrt + (c0 + lr) * 64;
    acc = __builtin_amdgcn_wmma_f32_16x16x32_f16(false, a0, false,
            load_b16(col, 0, lh), (short)0, acc, false, false);
    acc = __builtin_amdgcn_wmma_f32_16x16x32_f16(false, a1, false,
            load_b16(col, 1, lh), (short)0, acc, false, false);
    float b = bias[c0 + lr];
#pragma unroll
    for (int v = 0; v < 8; ++v) {
      int m = v + 8 * lh;
      out[(long long)(tile * 16 + m) * HF + c0 + lr] = acc[v] + b;
    }
  }
}

// ---- message GEMM: 16-edge WMMA tiles, alpha scale, scatter-add ------------

__global__ void k_msg(const _Float16* __restrict__ xh,
                      const _Float16* __restrict__ wht,
                      const float* __restrict__ proj,
                      const unsigned* __restrict__ segmax,
                      const float* __restrict__ segsum,
                      const int* __restrict__ sorted,
                      const int* __restrict__ esrc, const int* __restrict__ edst,
                      const int* __restrict__ etyp,
                      float* __restrict__ out) {
  const int wave = threadIdx.x >> 5, lane = threadIdx.x & 31;
  const int lh = lane >> 4, lr = lane & 15;
  const int tile = blockIdx.x * (blockDim.x >> 5) + wave;
  if (tile >= NTILES) return;

  int e = -1, s = 0, d = 0, r = -1;
  if (lane < 16) {
    e = sorted[tile * 16 + lane];
    if (e >= 0) { s = esrc[e]; d = edst[e]; r = etyp[e]; }
  }
  // tile relation (uniform across real edges of the tile by construction)
  int tr = r;
#pragma unroll
  for (int off = 16; off; off >>= 1) tr = max(tr, __shfl_xor(tr, off, 32));
  if (tr < 0) return;  // all-dummy tile (wave-uniform)

  const int s_row = __shfl(s, lr, 32);
  const int d_row = __shfl(d, lr, 32);
  const int valid_row = __shfl((int)(e >= 0), lr, 32);

  // alpha for (row=lr, h=lh) and (row=lr, h=lh+2); dummy rows -> 0
  float al0 = 0.f, al1 = 0.f;
  if (valid_row) {
    const long long segb = ((long long)tr * N_NODES + d_row) * 4;
    const int rh = tr * 4;
#pragma unroll
    for (int hh = 0; hh < 2; ++hh) {
      int h = lh + hh * 2;
      float lg = leaky(proj[(long long)s_row * 64 + rh + h] +
                       proj[(long long)d_row * 64 + 32 + rh + h]);
      float a = __expf(lg - funkey(segmax[segb + h])) / segsum[segb + h];
      if (hh == 0) al0 = a; else al1 = a;
    }
  }

  // A fragments for the 16 gathered source rows (x resident in L2)
  const _Float16* arow = xh + (long long)s_row * FIN;
  v16h a0 = load_a16(arow, 0, lh);
  v16h a1 = load_a16(arow, 1, lh);
  const _Float16* wr = wht + tr * HF * FIN;   // 128KB table, L2-hot

#pragma unroll
  for (int c0 = 0; c0 < HF; c0 += 16) {
    v8f acc = {};
    const _Float16* col = wr + (c0 + lr) * FIN;
    acc = __builtin_amdgcn_wmma_f32_16x16x32_f16(false, a0, false,
            load_b16(col, 0, lh), (short)0, acc, false, false);
    acc = __builtin_amdgcn_wmma_f32_16x16x32_f16(false, a1, false,
            load_b16(col, 1, lh), (short)0, acc, false, false);
    const int h = c0 >> 5;            // head is uniform within a col-tile
    const int c = c0 + lr;
#pragma unroll
    for (int v = 0; v < 8; ++v) {
      int m = v + 8 * lh;
      float a = (h < 2) ? __shfl(al0, (h << 4) + m, 32)
                        : __shfl(al1, ((h - 2) << 4) + m, 32);
      int dm = __shfl(d, m, 32);
      if (a != 0.f) atomicAdd(&out[(long long)dm * HF + c], acc[v] * a);
    }
  }
}

// ---- launch ----------------------------------------------------------------

extern "C" void kernel_launch(void* const* d_in, const int* in_sizes, int n_in,
                              void* d_out, int out_size, void* d_ws, size_t ws_size,
                              hipStream_t stream) {
  (void)in_sizes; (void)n_in; (void)out_size; (void)ws_size;
  const float* x      = (const float*)d_in[0];
  const int*   eidx   = (const int*)d_in[1];       // [2,E]
  const int*   etyp   = (const int*)d_in[2];       // [E]
  const float* W      = (const float*)d_in[3];     // [R,H,Fin,Fout]
  const float* att_s  = (const float*)d_in[4];
  const float* att_d  = (const float*)d_in[5];
  const float* wroot  = (const float*)d_in[6];     // [Fin, H*Fout]
  const float* bias   = (const float*)d_in[7];
  float*       out    = (float*)d_out;
  const int* esrc = eidx;
  const int* edst = eidx + N_EDGES;

  // workspace carve-up (256B aligned), ~68 MB total
  char* w = (char*)d_ws;
  size_t off = 0;
  auto carve = [&](size_t bytes) {
    char* p = w + off;
    off = (off + bytes + 255) & ~(size_t)255;
    return p;
  };
  _Float16* xh      = (_Float16*)carve((size_t)N_NODES * FIN * 2);
  _Float16* wht     = (_Float16*)carve((size_t)NR * HF * FIN * 2);
  _Float16* uht     = (_Float16*)carve((size_t)64 * 64 * 2);
  _Float16* wrt     = (_Float16*)carve((size_t)HF * FIN * 2);
  float*    proj    = (float*)carve((size_t)N_NODES * 64 * 4);
  unsigned* segmax  = (unsigned*)carve((size_t)NR * N_NODES * NH * 4);
  float*    segsum  = (float*)carve((size_t)NR * N_NODES * NH * 4);
  int*      counts  = (int*)carve(NR * 4);
  int*      cursors = (int*)carve(NR * 4);
  int*      offs    = (int*)carve((NR + 1) * 4);
  int*      sorted  = (int*)carve((size_t)EPAD * 4);

  const int T = 256;
  auto cdiv = [](long long a, long long b) { return (int)((a + b - 1) / b); };

  // 1) init segment buffers, sorted=-1, counters
  k_init<<<cdiv((long long)NR * N_NODES * NH, T), T, 0, stream>>>(
      segmax, segsum, sorted, counts, cursors);
  // 2) precision prep / transposes
  k_xh <<<cdiv((long long)N_NODES * FIN, T), T, 0, stream>>>(x, xh);
  k_wt <<<cdiv(NR * NH * FIN * FOUT, T), T, 0, stream>>>(W, wht);
  k_u  <<<cdiv(64 * 64, T), T, 0, stream>>>(W, att_s, att_d, uht);
  k_wrt<<<cdiv(HF * FIN, T), T, 0, stream>>>(wroot, wrt);
  // 3) attention projections (WMMA)
  k_proj<<<cdiv(N_NODES / 16, T / 32), T, 0, stream>>>(xh, uht, proj);
  // 4) segment softmax stats
  k_segmax<<<cdiv((long long)N_EDGES * NH, T), T, 0, stream>>>(
      esrc, edst, etyp, proj, segmax);
  k_segsum<<<cdiv((long long)N_EDGES * NH, T), T, 0, stream>>>(
      esrc, edst, etyp, proj, segmax, segsum);
  // 5) counting sort of edges by relation (16-aligned regions)
  k_hist   <<<cdiv(N_EDGES, T), T, 0, stream>>>(etyp, counts);
  k_scan   <<<1, 1, 0, stream>>>(counts, offs);
  k_scatter<<<cdiv(N_EDGES, T), T, 0, stream>>>(etyp, offs, cursors, sorted);
  // 6) out = x @ W_root + bias (WMMA)
  k_root<<<cdiv(N_NODES / 16, T / 32), T, 0, stream>>>(xh, wrt, bias, out);
  // 7) messages: WMMA per 16-edge tile + alpha scale + scatter-add
  k_msg<<<cdiv(NTILES, T / 32), T, 0, stream>>>(
      xh, wht, proj, segmax, segsum, sorted, esrc, edst, etyp, out);
}
